// Multi_modal_Fusion_Attention_4879082848541
// MI455X (gfx1250) — matrix operand order repaired and checked by reading, not compiled, not run on hardware
//
#include <hip/hip_runtime.h>

// ---------------------------------------------------------------------------
// Multi-modal fusion attention for MI455X (gfx1250, wave32, WMMA + TDM).
//   1. fp32 -> bf16 conversions (img, audio, 8 weight mats), W_bil transpose
//   2. 4 projections  Y = X @ W^T + b           (lin_kernel, fuse=0)
//   3. 4 fused attentions: softmax(QK^T)*NORM @ V, two-pass online softmax,
//      scores recomputed, P + V^T staged in LDS (attn_kernel)
//   4. 2 fused dual-linears: sigmoid((A1W1+b1 + A2W2+b2) * mult) (fuse=1)
//   5. bilinear + gating epilogue into d_out; W_o slabs double-buffered in
//      LDS via the Tensor Data Mover (bilinear_kernel)
// ---------------------------------------------------------------------------

typedef unsigned short u16;
typedef __bf16 bf16;
typedef bf16  v16bf  __attribute__((ext_vector_type(16)));
typedef float v8f    __attribute__((ext_vector_type(8)));
typedef u16   u16x8  __attribute__((ext_vector_type(8)));
typedef u16   u16x16 __attribute__((ext_vector_type(16)));

#define NB 8
#define NS 2048
#define ND 256
#define NTOK (NB * NS)
#define NORMC 0.0625f   // 1/sqrt(256)

static __device__ __forceinline__ u16 f2bf(float f) {
  unsigned u = __float_as_uint(f);
  u += 0x7fffu + ((u >> 16) & 1u);       // round-to-nearest-even
  return (u16)(u >> 16);
}
static __device__ __forceinline__ float bf2f(u16 h) {
  return __uint_as_float(((unsigned)h) << 16);
}
static __device__ __forceinline__ float sigf(float x) {
  return 1.0f / (1.0f + __expf(-x));
}
static __device__ __forceinline__ v8f wmma_bf(v16bf a, v16bf b, v8f c) {
  return __builtin_amdgcn_wmma_f32_16x16x32_bf16(false, a, false, b, (short)0, c,
                                                 false, false);
}

// A fragment (16x32 bf16): lane = row (l&15); lanes 0-15 hold K = k0+0..15,
// lanes 16-31 hold K = k0+16..31 (same k order as ld_b).
static __device__ __forceinline__ v16bf ld_a(const u16* base, int stride,
                                             int row0, int k0) {
  int l = (int)(threadIdx.x & 31u);
  const u16* p = base + (row0 + (l & 15)) * stride + k0 + ((l >> 4) << 4);
  return __builtin_bit_cast(v16bf, *(const u16x16*)(p));
}

// B fragment (32x16 bf16): lane = col (l&15); lanes 0-15 hold K = k0+0..15,
// lanes 16-31 hold K = k0+16..31. Column data contiguous in memory.
static __device__ __forceinline__ v16bf ld_b(const u16* base, int stride,
                                             int col0, int k0) {
  int l = (int)(threadIdx.x & 31u);
  const u16* p = base + (col0 + (l & 15)) * stride + k0 + ((l >> 4) << 4);
  return __builtin_bit_cast(v16bf, *(const u16x16*)(p));
}

// ---------------------------------------------------------------------------
// Tensor Data Mover: 2-D tile load (256 rows x 256 bf16, stride 256) -> LDS.
// This toolchain declares the 6-arg builtin:
//   (uint32x4 g0, int32x8 g1, int32x4 g2, int32x4 g3, int32x8 g4, i32 cpol)
#if __has_builtin(__builtin_amdgcn_tensor_load_to_lds)
#define HAVE_TDM 1
typedef unsigned int uint4v __attribute__((ext_vector_type(4)));
typedef int int8v __attribute__((ext_vector_type(8)));
typedef int int4v __attribute__((ext_vector_type(4)));

static __device__ __forceinline__ void tdm_load_slab(const u16* gsrc,
                                                     unsigned lds_off) {
  unsigned long long ga = (unsigned long long)(size_t)gsrc;
  uint4v g0;
  g0[0] = 1u;                                      // count=1, user mode
  g0[1] = lds_off;                                 // lds_addr (bytes)
  g0[2] = (unsigned)(ga & 0xffffffffu);            // global_addr[31:0]
  g0[3] = (unsigned)((ga >> 32) & 0x01ffffffu)     // global_addr[56:32]
          | (2u << 30);                            // type = 2 ("image")
  int8v g1;
  g1[0] = 0x00010000;                              // data_size = 2 bytes
  g1[1] = (int)(256u << 16);                       // tensor_dim0 = 256 (lo16)
  g1[2] = (int)(256u << 16);                       // tensor_dim1 = 256 (lo16)
  g1[3] = (int)(256u << 16);                       // tile_dim0 = 256
  g1[4] = 256;                                     // tile_dim1 = 256
  g1[5] = 256;                                     // tensor_dim0_stride = 256
  g1[6] = 0;
  g1[7] = 0;
  int4v z4 = {0, 0, 0, 0};                         // groups 2/3 unused (2-D)
  int8v z8 = {0, 0, 0, 0, 0, 0, 0, 0};
  __builtin_amdgcn_tensor_load_to_lds(g0, g1, z4, z4, z8, 0);
}
#else
#define HAVE_TDM 0
#endif

// ---------------------------------------------------------------------------
// elementwise fp32 -> bf16
__global__ void f32_to_bf16_kernel(const float* __restrict__ in,
                                   u16* __restrict__ out, int n) {
  int i = blockIdx.x * blockDim.x + threadIdx.x;
  if (i < n) out[i] = f2bf(in[i]);
}

// W_bil [o][i][j] fp32 -> Wt [o][j][i] bf16 (bilinear B-columns over i
// contiguous)
__global__ void wbil_transpose_kernel(const float* __restrict__ in,
                                      u16* __restrict__ out) {
  int idx = blockIdx.x * 256 + threadIdx.x;  // 2^24 elements
  int o = idx >> 16;
  int j = (idx >> 8) & 255;
  int i = idx & 255;
  out[idx] = f2bf(in[(o << 16) + (i << 8) + j]);
}

// ---------------------------------------------------------------------------
// Y = A1 @ W1^T + b1                       (fuse == 0)
// Y = sigmoid((A1W1+b1 + A2W2+b2) * mult)  (fuse == 1)
__global__ __launch_bounds__(256) void lin_kernel(
    const u16* __restrict__ A1, const u16* __restrict__ W1,
    const float* __restrict__ b1, const u16* __restrict__ A2,
    const u16* __restrict__ W2, const float* __restrict__ b2,
    const u16* __restrict__ mult, u16* __restrict__ out, int fuse) {
  int t0 = blockIdx.x * 64;
  int tid = threadIdx.x;
  int w = tid >> 5, l = tid & 31;
  int rg = w & 3, ch = w >> 2;
  int row_l = t0 + 16 * rg;

  v8f acc[8];
#pragma unroll
  for (int nt = 0; nt < 8; ++nt) { v8f z = {}; acc[nt] = z; }

#pragma unroll
  for (int s = 0; s < 8; ++s) {
    v16bf a = ld_a(A1, ND, row_l, 32 * s);
#pragma unroll
    for (int nt = 0; nt < 8; ++nt) {
      v16bf bb = ld_b(W1, ND, ch * 128 + nt * 16, 32 * s);
      acc[nt] = wmma_bf(a, bb, acc[nt]);
    }
  }
  if (fuse) {
#pragma unroll
    for (int s = 0; s < 8; ++s) {
      v16bf a = ld_a(A2, ND, row_l, 32 * s);
#pragma unroll
      for (int nt = 0; nt < 8; ++nt) {
        v16bf bb = ld_b(W2, ND, ch * 128 + nt * 16, 32 * s);
        acc[nt] = wmma_bf(a, bb, acc[nt]);
      }
    }
  }

#pragma unroll
  for (int nt = 0; nt < 8; ++nt) {
    int col = ch * 128 + nt * 16 + (l & 15);
    float bias = b1[col] + (fuse ? b2[col] : 0.0f);
#pragma unroll
    for (int v = 0; v < 8; ++v) {
      int row = row_l + v + 8 * (l >> 4);
      float val = acc[nt][v] + bias;
      if (fuse) {
        val = sigf(val * bf2f(mult[row * ND + col]));
      }
      out[row * ND + col] = f2bf(val);
    }
  }
}

// ---------------------------------------------------------------------------
// Fused attention: alpha = (softmax(Q K^T, axis=-1) * NORM) @ V  per batch.
static __device__ __forceinline__ void qk_tile(const v16bf* qa,
                                               const u16* __restrict__ Kb,
                                               int kcol0, v8f& s0, v8f& s1) {
  v8f z = {};
  s0 = z;
  s1 = z;
#pragma unroll
  for (int s = 0; s < 8; ++s) {
    v16bf b0 = ld_b(Kb, ND, kcol0, 32 * s);
    v16bf b1 = ld_b(Kb, ND, kcol0 + 16, 32 * s);
    s0 = wmma_bf(qa[s], b0, s0);
    s1 = wmma_bf(qa[s], b1, s1);
  }
}

__global__ __launch_bounds__(256) void attn_kernel(const u16* __restrict__ Q,
                                                   const u16* __restrict__ K,
                                                   const u16* __restrict__ V,
                                                   u16* __restrict__ alpha) {
  __shared__ float m_s[64];
  __shared__ float l_s[64];
  __shared__ float red[2][64];
  __shared__ alignas(32) u16 Pt[64 * 64];    // probs tile, bf16
  __shared__ alignas(32) u16 Vt[256 * 64];   // V^T tile: [d][t], bf16

  int b = blockIdx.y;
  int q0 = blockIdx.x * 64;
  const u16* Qb = Q + (size_t)b * NS * ND;
  const u16* Kb = K + (size_t)b * NS * ND;
  const u16* Vb = V + (size_t)b * NS * ND;

  int tid = threadIdx.x;
  int w = tid >> 5, l = tid & 31;
  int rg = w & 3, cg = w >> 2;
  int hi = l >> 4;

  if (tid < 64) { m_s[tid] = -3.0e38f; l_s[tid] = 0.0f; }
  __syncthreads();

  v16bf qa[8];
#pragma unroll
  for (int s = 0; s < 8; ++s) qa[s] = ld_a(Qb, ND, q0 + 16 * rg, 32 * s);

  // ---------------- pass 1: online row max + sum of exp -------------------
  for (int kt = 0; kt < NS / 64; ++kt) {
    int kcol0 = kt * 64 + cg * 32;
    v8f s0, s1;
    qk_tile(qa, Kb, kcol0, s0, s1);

#pragma unroll
    for (int v = 0; v < 8; ++v) {
      float x = fmaxf(s0[v], s1[v]);
#pragma unroll
      for (int mm = 8; mm >= 1; mm >>= 1) x = fmaxf(x, __shfl_xor(x, mm, 16));
      if ((l & 15) == 0) red[cg][16 * rg + v + 8 * hi] = x;
    }
    __syncthreads();
    if (tid < 64) {
      float tm = fmaxf(red[0][tid], red[1][tid]);
      float mo = m_s[tid];
      float mn = fmaxf(mo, tm);
      l_s[tid] = l_s[tid] * __expf(mo - mn);
      m_s[tid] = mn;
    }
    __syncthreads();
#pragma unroll
    for (int v = 0; v < 8; ++v) {
      int row = 16 * rg + v + 8 * hi;
      float mm = m_s[row];
      float e = __expf(s0[v] - mm) + __expf(s1[v] - mm);
#pragma unroll
      for (int sh = 8; sh >= 1; sh >>= 1) e += __shfl_xor(e, sh, 16);
      if ((l & 15) == 0) red[cg][row] = e;
    }
    __syncthreads();
    if (tid < 64) l_s[tid] += red[0][tid] + red[1][tid];
    __syncthreads();
  }

  // ---------------- pass 2: recompute scores, P @ V -----------------------
  v8f O[8];
#pragma unroll
  for (int nt = 0; nt < 8; ++nt) { v8f z = {}; O[nt] = z; }

  for (int kt = 0; kt < NS / 64; ++kt) {
    int kcol0 = kt * 64 + cg * 32;
    v8f s0, s1;
    qk_tile(qa, Kb, kcol0, s0, s1);

    __syncthreads();  // previous tile's Pt/Vt consumers done

#pragma unroll
    for (int v = 0; v < 8; ++v) {
      int row = 16 * rg + v + 8 * hi;
      float sc = NORMC / l_s[row];
      float mm = m_s[row];
      int col = l & 15;
      Pt[row * 64 + cg * 32 + col] = f2bf(__expf(s0[v] - mm) * sc);
      Pt[row * 64 + cg * 32 + 16 + col] = f2bf(__expf(s1[v] - mm) * sc);
    }

    // stage V^T tile into LDS: Vt[d][t] = V[kt*64+t][d]
    {
      int t = tid >> 2;
      int dg = (tid & 3) << 6;
      const u16* src = Vb + (size_t)(kt * 64 + t) * ND + dg;
#pragma unroll
      for (int c = 0; c < 8; ++c) {
        u16x8 vv = *(const u16x8*)(src + c * 8);
#pragma unroll
        for (int e = 0; e < 8; ++e) Vt[(dg + c * 8 + e) * 64 + t] = vv[e];
      }
    }
    __syncthreads();

#pragma unroll
    for (int s = 0; s < 2; ++s) {
      v16bf pa = ld_a(Pt, 64, 16 * rg, 32 * s);
#pragma unroll
      for (int nt = 0; nt < 8; ++nt) {
        v16bf vb = ld_b(Vt, 64, cg * 128 + nt * 16, 32 * s);
        O[nt] = wmma_bf(pa, vb, O[nt]);
      }
    }
  }

#pragma unroll
  for (int nt = 0; nt < 8; ++nt) {
    int col = cg * 128 + nt * 16 + (l & 15);
#pragma unroll
    for (int v = 0; v < 8; ++v) {
      int row = q0 + 16 * rg + v + 8 * hi;
      alpha[((size_t)b * NS + row) * ND + col] = f2bf(O[nt][v]);
    }
  }
}

// ---------------------------------------------------------------------------
// y[t,o] = sum_{i,j} x1[t,i] W[o,i,j] x2[t,j];  out = tc*sig(y)*img+(1-sig)*aud
// W_o slabs (128 KB, [j][i] bf16) double-buffered in LDS via TDM; wave 0
// issues tensor_load_to_lds for o+1 while all waves compute o from LDS.
__global__ __launch_bounds__(256) void bilinear_kernel(
    const u16* __restrict__ X1, const u16* __restrict__ X2,
    const u16* __restrict__ Wt, const float* __restrict__ img,
    const float* __restrict__ aud, const float* __restrict__ tc,
    float* __restrict__ out) {
  extern __shared__ char smem_raw[];
  u16* buf0 = (u16*)smem_raw;                       // 128 KB
  u16* buf1 = (u16*)(smem_raw + 131072);            // 128 KB
  float* red = (float*)(smem_raw + 262144);         // [2][64]

  int t0 = blockIdx.x * 64;
  int tid = threadIdx.x;
  int w = tid >> 5, l = tid & 31;
  int rg = w & 3, jh = w >> 2;
  int hi = l >> 4;

  // preload X1 A-fragments (16 tokens x K=256)
  v16bf a[8];
#pragma unroll
  for (int s = 0; s < 8; ++s) a[s] = ld_a(X1, ND, t0 + 16 * rg, 32 * s);

  // preload X2 in C-fragment layout (lane=col, vgpr=row) as fp32
  float x2[8][8];
#pragma unroll
  for (int nt = 0; nt < 8; ++nt) {
    int col = jh * 128 + nt * 16 + (l & 15);
#pragma unroll
    for (int v = 0; v < 8; ++v) {
      int row = t0 + 16 * rg + v + 8 * hi;
      x2[nt][v] = bf2f(X2[row * ND + col]);
    }
  }
  float tcv = tc[0];

#if HAVE_TDM
  if (w == 0) tdm_load_slab(Wt, (unsigned)(size_t)(void*)buf0);
#endif

  for (int o = 0; o < 256; ++o) {
    u16* cur = (o & 1) ? buf1 : buf0;
#if HAVE_TDM
    if (w == 0) {
      if (o < 255) {
        u16* nxt = (o & 1) ? buf0 : buf1;
        tdm_load_slab(Wt + ((size_t)(o + 1) << 16),
                      (unsigned)(size_t)(void*)nxt);
        __builtin_amdgcn_s_wait_tensorcnt(1);   // current slab landed
      } else {
        __builtin_amdgcn_s_wait_tensorcnt(0);
      }
    }
    __syncthreads();   // slab visible to all waves
#else
    {
      const u16* gsrc = Wt + ((size_t)o << 16);
      for (int t = tid; t < 8192; t += 256)
        *(u16x8*)(cur + t * 8) = *(const u16x8*)(gsrc + t * 8);
    }
    __syncthreads();
#endif

    float part[8] = {0.f, 0.f, 0.f, 0.f, 0.f, 0.f, 0.f, 0.f};
#pragma unroll
    for (int nt = 0; nt < 8; ++nt) {
      v8f g = {};
#pragma unroll
      for (int s = 0; s < 8; ++s) {
        v16bf bb = ld_b(cur, ND, jh * 128 + nt * 16, 32 * s);
        g = wmma_bf(a[s], bb, g);
      }
#pragma unroll
      for (int v = 0; v < 8; ++v) part[v] += g[v] * x2[nt][v];
    }
    // row-sum across this wave's 128 j columns (16 lanes per half)
#pragma unroll
    for (int v = 0; v < 8; ++v) {
      float x = part[v];
#pragma unroll
      for (int mm = 8; mm >= 1; mm >>= 1) x += __shfl_xor(x, mm, 16);
      if ((l & 15) == 0) red[jh * 64 + 16 * rg + v + 8 * hi] = x;
    }
    __syncthreads();
    if (tid < 64) {
      float y = red[tid] + red[64 + tid];
      float j = sigf(y);
      size_t idx = (size_t)(t0 + tid) * ND + o;
      out[idx] = tcv * j * img[idx] + (1.0f - j) * aud[idx];
    }
    __syncthreads();   // end-of-iter: safe to overwrite buf[o&1]
  }
}

// ---------------------------------------------------------------------------
extern "C" void kernel_launch(void* const* d_in, const int* in_sizes, int n_in,
                              void* d_out, int out_size, void* d_ws,
                              size_t ws_size, hipStream_t stream) {
  (void)in_sizes; (void)n_in; (void)out_size; (void)ws_size;

  const float* img   = (const float*)d_in[0];
  const float* aud   = (const float*)d_in[1];
  const float* Wf[8] = {(const float*)d_in[2],  (const float*)d_in[4],
                        (const float*)d_in[6],  (const float*)d_in[8],
                        (const float*)d_in[10], (const float*)d_in[12],
                        (const float*)d_in[14], (const float*)d_in[16]};
  const float* bf_[8] = {(const float*)d_in[3],  (const float*)d_in[5],
                         (const float*)d_in[7],  (const float*)d_in[9],
                         (const float*)d_in[11], (const float*)d_in[13],
                         (const float*)d_in[15], (const float*)d_in[17]};
  const float* W_bil = (const float*)d_in[18];
  const float* t_c   = (const float*)d_in[19];
  float* out = (float*)d_out;

  char* p = (char*)d_ws;
  auto carve = [&](size_t bytes) -> void* {
    void* r = (void*)p;
    p += (bytes + 255) & ~(size_t)255;
    return r;
  };
  const size_t tokb = (size_t)NTOK * ND * sizeof(u16);  // 8 MB
  u16* imgB = (u16*)carve(tokb);
  u16* audB = (u16*)carve(tokb);
  u16* Qv   = (u16*)carve(tokb);
  u16* Qa   = (u16*)carve(tokb);
  u16* Kv   = (u16*)carve(tokb);
  u16* Ka   = (u16*)carve(tokb);
  u16* Ava  = (u16*)carve(tokb);
  u16* Av   = (u16*)carve(tokb);
  u16* Aav  = (u16*)carve(tokb);
  u16* Aa   = (u16*)carve(tokb);
  u16* CKva = (u16*)carve(tokb);
  u16* CKav = (u16*)carve(tokb);
  u16* Wb[8];
  for (int i = 0; i < 8; ++i) Wb[i] = (u16*)carve((size_t)ND * ND * sizeof(u16));
  u16* WtB = (u16*)carve((size_t)ND * ND * ND * sizeof(u16));  // 33.5 MB

  const int TPB = 256;
  const int nTok = NTOK * ND;

  // 1. conversions
  f32_to_bf16_kernel<<<(nTok + TPB - 1) / TPB, TPB, 0, stream>>>(img, imgB, nTok);
  f32_to_bf16_kernel<<<(nTok + TPB - 1) / TPB, TPB, 0, stream>>>(aud, audB, nTok);
  for (int i = 0; i < 8; ++i)
    f32_to_bf16_kernel<<<(ND * ND + TPB - 1) / TPB, TPB, 0, stream>>>(
        Wf[i], Wb[i], ND * ND);
  wbil_transpose_kernel<<<(1 << 24) / TPB, TPB, 0, stream>>>(W_bil, WtB);

  // 2. projections
  lin_kernel<<<NTOK / 64, TPB, 0, stream>>>(imgB, Wb[0], bf_[0], nullptr,
                                            nullptr, nullptr, nullptr, Qv, 0);
  lin_kernel<<<NTOK / 64, TPB, 0, stream>>>(audB, Wb[1], bf_[1], nullptr,
                                            nullptr, nullptr, nullptr, Qa, 0);
  lin_kernel<<<NTOK / 64, TPB, 0, stream>>>(imgB, Wb[2], bf_[2], nullptr,
                                            nullptr, nullptr, nullptr, Kv, 0);
  lin_kernel<<<NTOK / 64, TPB, 0, stream>>>(audB, Wb[3], bf_[3], nullptr,
                                            nullptr, nullptr, nullptr, Ka, 0);

  // 3. fused attentions
  dim3 agrid(NS / 64, NB);
  attn_kernel<<<agrid, TPB, 0, stream>>>(Qv, Ka, audB, Ava);
  attn_kernel<<<agrid, TPB, 0, stream>>>(Qv, Kv, imgB, Av);
  attn_kernel<<<agrid, TPB, 0, stream>>>(Qa, Kv, imgB, Aav);
  attn_kernel<<<agrid, TPB, 0, stream>>>(Qa, Ka, audB, Aa);

  // 4. gated channels
  lin_kernel<<<NTOK / 64, TPB, 0, stream>>>(Ava, Wb[4], bf_[4], Av, Wb[5],
                                            bf_[5], Av, CKva, 1);
  lin_kernel<<<NTOK / 64, TPB, 0, stream>>>(Aav, Wb[6], bf_[6], Aa, Wb[7],
                                            bf_[7], Aa, CKav, 1);

  // 5. bilinear + output gating (2 x 128 KB LDS slabs + reduction scratch)
  const size_t bil_smem = 262144 + 512;
  bilinear_kernel<<<NTOK / 64, TPB, bil_smem, stream>>>(CKva, CKav, WtB, img,
                                                        aud, t_c, out);
}
